// LDM_5815385719340
// MI455X (gfx1250) — compile-verified, matrix operand order checked by reading, not run on hardware
//
#include <hip/hip_runtime.h>

#define DMODEL 512
#define NB 32          // batch
#define LPV 224        // padded vis-length (196 -> 224, multiple of 32)
#define LXX 256        // x/text length
#define LVVALID 196    // valid vis rows

typedef __attribute__((ext_vector_type(16))) __bf16 v16bf;
typedef __attribute__((ext_vector_type(8)))  float  v8f;
typedef __attribute__((ext_vector_type(4)))  unsigned int u32x4;
typedef __attribute__((ext_vector_type(2)))  unsigned int u32x2;

union FragBF { v16bf v; unsigned short u[16]; u32x4 q[2]; };

__device__ __forceinline__ unsigned short f2bf(float f) {
  union { __bf16 h; unsigned short u; } c;
  c.h = (__bf16)f;            // hardware v_cvt f32->bf16
  return c.u;
}

__device__ __forceinline__ unsigned int pack2bf(float a, float b) {
  union { __bf16 h[2]; unsigned int u; } c;
  c.h[0] = (__bf16)a;
  c.h[1] = (__bf16)b;         // lowers to v_cvt_pk_bf16_f32
  return c.u;
}

// A-operand fragment (16x32) for V_WMMA_F32_16X16X32_BF16 from row-major f32.
// lane l holds row rbase+(l&15); elems 0..7 -> k = kbase+(l>>4)*8+i,
// elems 8..15 -> k = kbase+16+(l>>4)*8+(i-8)   (per ISA 16-bit A layout)
__device__ __forceinline__ v16bf load_frag_f32(const float* __restrict__ base,
                                               int stride, int rbase, int kbase,
                                               int lane, int rmax) {
  int r = rbase + (lane & 15);
  r = r > rmax ? rmax : r;
  const float* p = base + (size_t)r * stride + kbase + ((lane >> 4) << 3);
  v16bf v;
#pragma unroll
  for (int i = 0; i < 8; ++i) v[i] = (__bf16)p[i];
#pragma unroll
  for (int i = 0; i < 8; ++i) v[8 + i] = (__bf16)p[16 + i];
  return v;
}

// Same A layout but source already bf16 row-major (softmax probabilities).
__device__ __forceinline__ v16bf load_frag_bf16(const unsigned short* __restrict__ base,
                                                int stride, int rbase, int kbase,
                                                int lane, int rmax) {
  int r = rbase + (lane & 15);
  r = r > rmax ? rmax : r;
  const unsigned short* p = base + (size_t)r * stride + kbase + ((lane >> 4) << 3);
  FragBF f;
  f.q[0] = *(const u32x4*)p;
  f.q[1] = *(const u32x4*)(p + 16);
  return f.v;
}

// B-operand fragment from pre-packed layout: 32 lanes x 16 bf16 contiguous.
__device__ __forceinline__ v16bf load_frag_packed(const unsigned short* __restrict__ pk,
                                                  int lane) {
  FragBF f;
  const u32x4* p = (const u32x4*)(pk + (lane << 4));
  f.q[0] = p[0];
  f.q[1] = p[1];
  return f.v;
}

// Fragment from LDS (already in fragment layout): two ds_load_b128 per lane.
__device__ __forceinline__ v16bf load_frag_lds(const unsigned short* p) {
  FragBF f;
  f.q[0] = *(const u32x4*)p;
  f.q[1] = *(const u32x4*)(p + 8);
  return f.v;
}

__device__ __forceinline__ v8f vzero8() {
  v8f z = {0.f, 0.f, 0.f, 0.f, 0.f, 0.f, 0.f, 0.f};
  return z;
}

// ---------------------------------------------------------------------------
// Pack a row-major [K,N] f32 weight into bf16 B-fragment layout:
// fragment (chunk=k>>5, tile_n=n>>4) is 512 contiguous bf16:
//   lane = (n&15) + 16*((k>>3)&1) ; idx = ((k>>4)&1)*8 + (k&7)
// ---------------------------------------------------------------------------
__global__ void pack_w_kernel(const float* __restrict__ W,
                              unsigned short* __restrict__ dst,
                              int Kd, int N) {
  size_t i = (size_t)blockIdx.x * 256 + threadIdx.x;
  if (i >= (size_t)Kd * N) return;
  int k = (int)(i / N);
  int n = (int)(i - (size_t)k * N);
  int lanep = (n & 15) + (((k >> 3) & 1) << 4);
  int idx = (((k >> 4) & 1) << 3) | (k & 7);
  size_t off = ((size_t)((k >> 5) * (N >> 4) + (n >> 4)) << 9) + (lanep << 4) + idx;
  dst[off] = f2bf(W[i]);
}

// Pack activations [B, rows, 512] f32 -> per-batch B-fragment layout (bf16).
__global__ void pack_act_kernel(const float* __restrict__ src,
                                unsigned short* __restrict__ dst,
                                int rows, int Bn) {
  size_t i = (size_t)blockIdx.x * 256 + threadIdx.x;
  size_t total = (size_t)Bn * rows * DMODEL;
  if (i >= total) return;
  int n = (int)(i & 511);
  size_t t = i >> 9;
  int k = (int)(t % rows);
  int b = (int)(t / rows);
  int lanep = (n & 15) + (((k >> 3) & 1) << 4);
  int idx = (((k >> 4) & 1) << 3) | (k & 7);
  size_t off = (size_t)b * rows * DMODEL +
               ((size_t)((k >> 5) * (DMODEL >> 4) + (n >> 4)) << 9) + (lanep << 4) + idx;
  dst[off] = f2bf(src[i]);
}

// Copy [B,196,512] -> [B,224,512] (pad rows pre-zeroed by memset).
__global__ void pad_copy_kernel(const float* __restrict__ src, float* __restrict__ dst,
                                int rowsSrc, int rowsDst, int Bn) {
  size_t i = (size_t)blockIdx.x * 256 + threadIdx.x;
  size_t total = (size_t)Bn * rowsSrc * DMODEL;
  if (i >= total) return;
  int c = (int)(i & 511);
  size_t t = i >> 9;
  int r = (int)(t % rowsSrc);
  int b = (int)(t / rowsSrc);
  dst[((size_t)b * rowsDst + r) * DMODEL + c] = src[i];
}

// ---------------------------------------------------------------------------
// LayerNorm over D=512, one wave per row.
// ---------------------------------------------------------------------------
__global__ __launch_bounds__(256) void ln_kernel(const float* __restrict__ src,
                                                 float* __restrict__ dst,
                                                 const float* __restrict__ gamma,
                                                 const float* __restrict__ beta,
                                                 int srcRows, int dstRows, int Bn) {
  int wid = blockIdx.x * 8 + (threadIdx.x >> 5);
  int total = Bn * dstRows;
  if (wid >= total) return;
  int lane = threadIdx.x & 31;
  int r = wid % dstRows;
  int b = wid / dstRows;
  int sr = r < srcRows ? r : srcRows - 1;
  const float* x = src + ((size_t)b * srcRows + sr) * DMODEL;
  float v[16];
  float s = 0.f;
#pragma unroll
  for (int i = 0; i < 16; ++i) { v[i] = x[lane + i * 32]; s += v[i]; }
#pragma unroll
  for (int off = 16; off > 0; off >>= 1) s += __shfl_xor(s, off, 32);
  float mean = s * (1.0f / DMODEL);
  float q = 0.f;
#pragma unroll
  for (int i = 0; i < 16; ++i) { float d = v[i] - mean; q += d * d; }
#pragma unroll
  for (int off = 16; off > 0; off >>= 1) q += __shfl_xor(q, off, 32);
  float rstd = rsqrtf(q * (1.0f / DMODEL) + 1e-6f);
  float* o = dst + ((size_t)b * dstRows + r) * DMODEL;
#pragma unroll
  for (int i = 0; i < 16; ++i) {
    int c = lane + i * 32;
    o[c] = gamma[c] * (v[i] - mean) * rstd + beta[c];
  }
}

// ---------------------------------------------------------------------------
// GEMM: out[M,N] = act(A[M,K] @ Wpacked + bias) (+res1) (+res2)
// block = 8 waves (2x4), tile 64x128; M % 64 == 0, N % 128 == 0, K % 32 == 0.
// A tile staged per-block in LDS (bf16, fragment layout, double-buffered) so
// the 4 n-waves share one global fetch + convert of each A row.
// RELU/R1/R2 are compile-time so the epilogue is branch-free.
// ---------------------------------------------------------------------------
template <int RELU, int R1, int R2>
__global__ __launch_bounds__(256) void gemm_kernel(const float* __restrict__ A,
                                                   const unsigned short* __restrict__ WP,
                                                   const float* __restrict__ bias,
                                                   const float* __restrict__ res1,
                                                   const float* __restrict__ res2,
                                                   float* __restrict__ out,
                                                   int M, int N, int K) {
  __shared__ unsigned short ldsA[2][2048];   // 2 x (64x32 bf16) = 2 x 4KB
  int tid = threadIdx.x, lane = tid & 31, wave = tid >> 5;
  int wm = wave >> 2, wn = wave & 3;
  int mBlk = blockIdx.y * 64;
  int n0 = blockIdx.x * 128 + wn * 32;
  int nt16 = N >> 4;
  int nTile = n0 >> 4;
  int NC = K >> 5;

  // Per-thread staging slots: two groups of 4 consecutive k, rows mA and mB = mA+32.
  int f0 = tid, f1 = tid + 256;
  int mA = f0 >> 3, kqA = (f0 & 7) << 2;
  int mB = f1 >> 3, kqB = (f1 & 7) << 2;
  const float* gA = A + (size_t)(mBlk + mA) * K + kqA;
  const float* gB = A + (size_t)(mBlk + mB) * K + kqB;
  int loA = (mA >> 4) * 512 + ((mA & 15) + (((kqA >> 3) & 1) << 4)) * 16 +
            ((((kqA >> 4) & 1) << 3) | (kqA & 7));
  int loB = (mB >> 4) * 512 + ((mB & 15) + (((kqB >> 3) & 1) << 4)) * 16 +
            ((((kqB >> 4) & 1) << 3) | (kqB & 7));

  auto stage = [&](int c, int buf) {
    const float* p0 = gA + (c << 5);
    const float* p1 = gB + (c << 5);
    __builtin_prefetch(p0 + 32, 0, 0);
    __builtin_prefetch(p1 + 32, 0, 0);
    u32x2 w0, w1;
    w0[0] = pack2bf(p0[0], p0[1]); w0[1] = pack2bf(p0[2], p0[3]);
    w1[0] = pack2bf(p1[0], p1[1]); w1[1] = pack2bf(p1[2], p1[3]);
    *(u32x2*)(&ldsA[buf][loA]) = w0;
    *(u32x2*)(&ldsA[buf][loB]) = w1;
  };

  v8f acc[2][2];
  acc[0][0] = vzero8(); acc[0][1] = vzero8();
  acc[1][0] = vzero8(); acc[1][1] = vzero8();

  stage(0, 0);
  __syncthreads();
  for (int c = 0; c < NC; ++c) {
    int cur = c & 1;
    if (c + 1 < NC) stage(c + 1, cur ^ 1);
    const unsigned short* aB = ldsA[cur];
    v16bf a0 = load_frag_lds(aB + (wm * 2 + 0) * 512 + (lane << 4));
    v16bf a1 = load_frag_lds(aB + (wm * 2 + 1) * 512 + (lane << 4));
    v16bf b0 = load_frag_packed(WP + ((size_t)(c * nt16 + nTile) << 9), lane);
    v16bf b1 = load_frag_packed(WP + ((size_t)(c * nt16 + nTile + 1) << 9), lane);
    acc[0][0] = __builtin_amdgcn_wmma_f32_16x16x32_bf16(false, a0, false, b0, (short)0, acc[0][0], false, false);
    acc[0][1] = __builtin_amdgcn_wmma_f32_16x16x32_bf16(false, a0, false, b1, (short)0, acc[0][1], false, false);
    acc[1][0] = __builtin_amdgcn_wmma_f32_16x16x32_bf16(false, a1, false, b0, (short)0, acc[1][0], false, false);
    acc[1][1] = __builtin_amdgcn_wmma_f32_16x16x32_bf16(false, a1, false, b1, (short)0, acc[1][1], false, false);
    __syncthreads();
  }

  int rowHalf = (lane >> 4) << 3;
  int colLane = lane & 15;
#pragma unroll
  for (int am = 0; am < 2; ++am)
#pragma unroll
    for (int bn = 0; bn < 2; ++bn) {
      int col = n0 + bn * 16 + colLane;
      float bv = bias[col];
#pragma unroll
      for (int j = 0; j < 8; ++j) {
        int row = mBlk + wm * 32 + am * 16 + rowHalf + j;
        size_t off = (size_t)row * N + col;
        float val = acc[am][bn][j] + bv;
        if (RELU) val = fmaxf(val, 0.0f);
        if (R1) val += res1[off];
        if (R2) val += res2[off];
        out[off] = val;
      }
    }
}

// ---------------------------------------------------------------------------
// Batched attention scores: S[z, Lq, Lk] = scale * Q_h @ K_h^T
// z = b*H + h ; Q batch stride Lq*512, K batch stride Lk*512, head cols h*headDim.
// B-operand of Q@K^T is K^T, which from row-major K loads exactly like an
// A-fragment (lane&15 selects the key row) -> no repacking needed.
// ---------------------------------------------------------------------------
__global__ __launch_bounds__(256) void score_kernel(const float* __restrict__ Q,
                                                    const float* __restrict__ Km,
                                                    float* __restrict__ S,
                                                    int Lq, int Lk, int headDim,
                                                    int H, float scale) {
  int tid = threadIdx.x, lane = tid & 31, wave = tid >> 5;
  int wm = wave >> 2, wn = wave & 3;
  int z = blockIdx.z, b = z / H, h = z - b * H;
  int m0 = blockIdx.y * 64 + wm * 32;
  int n0 = blockIdx.x * 128 + wn * 32;
  const float* qb = Q + (size_t)b * Lq * DMODEL + h * headDim;
  const float* kb = Km + (size_t)b * Lk * DMODEL + h * headDim;
  float* sb = S + (size_t)z * Lq * Lk;
  v8f acc[2][2];
  acc[0][0] = vzero8(); acc[0][1] = vzero8();
  acc[1][0] = vzero8(); acc[1][1] = vzero8();
#pragma unroll 2
  for (int c = 0; c < (headDim >> 5); ++c) {
    v16bf a0 = load_frag_f32(qb, DMODEL, m0, c << 5, lane, Lq - 1);
    v16bf a1 = load_frag_f32(qb, DMODEL, m0 + 16, c << 5, lane, Lq - 1);
    v16bf b0 = load_frag_f32(kb, DMODEL, n0, c << 5, lane, Lk - 1);
    v16bf b1 = load_frag_f32(kb, DMODEL, n0 + 16, c << 5, lane, Lk - 1);
    acc[0][0] = __builtin_amdgcn_wmma_f32_16x16x32_bf16(false, a0, false, b0, (short)0, acc[0][0], false, false);
    acc[0][1] = __builtin_amdgcn_wmma_f32_16x16x32_bf16(false, a0, false, b1, (short)0, acc[0][1], false, false);
    acc[1][0] = __builtin_amdgcn_wmma_f32_16x16x32_bf16(false, a1, false, b0, (short)0, acc[1][0], false, false);
    acc[1][1] = __builtin_amdgcn_wmma_f32_16x16x32_bf16(false, a1, false, b1, (short)0, acc[1][1], false, false);
  }
  int rowHalf = (lane >> 4) << 3;
  int colLane = lane & 15;
#pragma unroll
  for (int am = 0; am < 2; ++am)
#pragma unroll
    for (int bn = 0; bn < 2; ++bn) {
      int col = n0 + bn * 16 + colLane;
#pragma unroll
      for (int j = 0; j < 8; ++j) {
        int row = m0 + am * 16 + rowHalf + j;
        if (row < Lq && col < Lk) sb[(size_t)row * Lk + col] = acc[am][bn][j] * scale;
      }
    }
}

// ---------------------------------------------------------------------------
// Row softmax with column masking (keys >= validCols get exactly 0), bf16 out.
// One wave per row; Lk % 32 == 0, Lk <= 256.
// ---------------------------------------------------------------------------
__global__ __launch_bounds__(256) void softmax_kernel(const float* __restrict__ S,
                                                      unsigned short* __restrict__ P,
                                                      int Lk, int validCols, int nRows) {
  int wid = blockIdx.x * 8 + (threadIdx.x >> 5);
  if (wid >= nRows) return;
  int lane = threadIdx.x & 31;
  const float* row = S + (size_t)wid * Lk;
  int nc = Lk >> 5;
  float x[8];
  float mx = -3.0e38f;
  for (int i = 0; i < nc; ++i) {
    int c = lane + (i << 5);
    x[i] = (c < validCols) ? row[c] : -3.0e38f;
    mx = fmaxf(mx, x[i]);
  }
#pragma unroll
  for (int off = 16; off > 0; off >>= 1) mx = fmaxf(mx, __shfl_xor(mx, off, 32));
  float s = 0.f;
  for (int i = 0; i < nc; ++i) {
    int c = lane + (i << 5);
    float e = (c < validCols) ? __expf(x[i] - mx) : 0.0f;
    x[i] = e;
    s += e;
  }
#pragma unroll
  for (int off = 16; off > 0; off >>= 1) s += __shfl_xor(s, off, 32);
  float inv = 1.0f / s;
  unsigned short* pr = P + (size_t)wid * Lk;
  for (int i = 0; i < nc; ++i) pr[lane + (i << 5)] = f2bf(x[i] * inv);
}

// ---------------------------------------------------------------------------
// O[b, Lq, h*headW + n] = P[z, Lq, Lk] @ Vpacked[b]  (batched over z = b*H+h)
// ---------------------------------------------------------------------------
__global__ __launch_bounds__(256) void pav_kernel(const unsigned short* __restrict__ P,
                                                  const unsigned short* __restrict__ Vp,
                                                  float* __restrict__ out,
                                                  int Lq, int Lk, int headW, int H) {
  int tid = threadIdx.x, lane = tid & 31, wave = tid >> 5;
  int wm = wave >> 2, wn = wave & 3;
  int z = blockIdx.z, b = z / H, h = z - b * H;
  int m0 = blockIdx.y * 64 + wm * 32;
  int n0 = blockIdx.x * 128 + wn * 32;
  if (n0 >= headW) return;
  const unsigned short* pb = P + (size_t)z * Lq * Lk;
  const unsigned short* vb = Vp + (size_t)b * Lk * DMODEL;
  int tngBase = (h * headW + n0) >> 4;
  v8f acc[2][2];
  acc[0][0] = vzero8(); acc[0][1] = vzero8();
  acc[1][0] = vzero8(); acc[1][1] = vzero8();
#pragma unroll 2
  for (int c = 0; c < (Lk >> 5); ++c) {
    v16bf a0 = load_frag_bf16(pb, Lk, m0, c << 5, lane, Lq - 1);
    v16bf a1 = load_frag_bf16(pb, Lk, m0 + 16, c << 5, lane, Lq - 1);
    v16bf b0 = load_frag_packed(vb + ((size_t)(c * (DMODEL >> 4) + tngBase) << 9), lane);
    v16bf b1 = load_frag_packed(vb + ((size_t)(c * (DMODEL >> 4) + tngBase + 1) << 9), lane);
    acc[0][0] = __builtin_amdgcn_wmma_f32_16x16x32_bf16(false, a0, false, b0, (short)0, acc[0][0], false, false);
    acc[0][1] = __builtin_amdgcn_wmma_f32_16x16x32_bf16(false, a0, false, b1, (short)0, acc[0][1], false, false);
    acc[1][0] = __builtin_amdgcn_wmma_f32_16x16x32_bf16(false, a1, false, b0, (short)0, acc[1][0], false, false);
    acc[1][1] = __builtin_amdgcn_wmma_f32_16x16x32_bf16(false, a1, false, b1, (short)0, acc[1][1], false, false);
  }
  int rowHalf = (lane >> 4) << 3;
  int colLane = lane & 15;
#pragma unroll
  for (int am = 0; am < 2; ++am)
#pragma unroll
    for (int bn = 0; bn < 2; ++bn) {
      int col_l = n0 + bn * 16 + colLane;
      if (col_l >= headW) continue;
      int gcol = h * headW + col_l;
#pragma unroll
      for (int j = 0; j < 8; ++j) {
        int row = m0 + am * 16 + rowHalf + j;
        if (row < Lq)
          out[(size_t)b * Lq * DMODEL + (size_t)row * DMODEL + gcol] = acc[am][bn][j];
      }
    }
}

// ---------------------------------------------------------------------------
extern "C" void kernel_launch(void* const* d_in, const int* in_sizes, int n_in,
                              void* d_out, int out_size, void* d_ws, size_t ws_size,
                              hipStream_t stream) {
  const float* x        = (const float*)d_in[0];   // [32,256,512]
  const float* text     = (const float*)d_in[1];   // [32,256,512]
  const float* vis      = (const float*)d_in[2];   // [32,196,512]
  const float* ft_aw    = (const float*)d_in[3];   // [4,512,512]
  const float* ft_ab    = (const float*)d_in[4];   // [4,512]
  const float* ft_w1    = (const float*)d_in[5];   // [512,2048]
  const float* ft_b1    = (const float*)d_in[6];
  const float* ft_w2    = (const float*)d_in[7];   // [2048,512]
  const float* ft_b2    = (const float*)d_in[8];
  const float* ft_ln    = (const float*)d_in[9];   // [2,2,512]
  const float* fv_aw    = (const float*)d_in[10];
  const float* fv_ab    = (const float*)d_in[11];
  const float* fv_w1    = (const float*)d_in[12];
  const float* fv_b1    = (const float*)d_in[13];
  const float* fv_w2    = (const float*)d_in[14];
  const float* fv_b2    = (const float*)d_in[15];
  const float* fv_ln    = (const float*)d_in[16];
  const float* ldm_ln   = (const float*)d_in[17];  // [2,512]
  const float* af_w     = (const float*)d_in[18];  // [2,2,512,512]
  const float* af_b     = (const float*)d_in[19];  // [2,2,512]
  float* outp = (float*)d_out;

  // ---- workspace layout -----------------------------------------------
  const size_t SZ_WP  = 7340032ULL * 2;                     // packed weights bf16
  const size_t SZ_VIS = (size_t)NB * LPV * DMODEL * 4;      // vis padded
  const size_t SZ_ACT = (size_t)NB * 256 * DMODEL * 4;      // generic act slab
  const size_t SZ_S   = (size_t)NB * 8 * LPV * 256 * 4;     // scores / ffn hidden
  const size_t SZ_P   = SZ_S / 2;                           // probs bf16
  const size_t SZ_PK  = (size_t)NB * 256 * DMODEL * 2;      // packed V bf16

  char* ws = (char*)d_ws;
  size_t off = 0;
  unsigned short* WP   = (unsigned short*)(ws + off); off += SZ_WP;
  float* VISP          = (float*)(ws + off);          off += SZ_VIS;
  float* LNb           = (float*)(ws + off);          off += SZ_ACT;
  float* Qb            = (float*)(ws + off);          off += SZ_ACT;
  float* Kb            = (float*)(ws + off);          off += SZ_ACT;
  float* Vb            = (float*)(ws + off);          off += SZ_ACT;
  float* M1b           = (float*)(ws + off);          off += SZ_ACT;
  float* Sb            = (float*)(ws + off);          off += SZ_S;
  unsigned short* Pb   = (unsigned short*)(ws + off); off += SZ_P;
  unsigned short* PKb  = (unsigned short*)(ws + off); off += SZ_PK;
  size_t used = off;
  hipMemsetAsync(d_ws, 0, used <= ws_size ? used : ws_size, stream);

  // ---- packed weight element offsets ----------------------------------
  const size_t pWq_ft = 0,        pWk_ft = 262144, pWv_ft = 524288, pWo_ft = 786432;
  const size_t pW1_ft = 1048576,  pW2_ft = 2097152;
  const size_t pWq_fv = 3145728,  pWk_fv = 3407872, pWv_fv = 3670016, pWo_fv = 3932160;
  const size_t pW1_fv = 4194304,  pW2_fv = 5242880;
  const size_t pW00   = 6291456,  pW01   = 6553600, pW10   = 6815744, pW11 = 7077888;

  struct PWDesc { const float* src; size_t dst; int K; int N; };
  PWDesc packs[16] = {
    {ft_aw + 0 * 262144, pWq_ft, 512, 512}, {ft_aw + 1 * 262144, pWk_ft, 512, 512},
    {ft_aw + 2 * 262144, pWv_ft, 512, 512}, {ft_aw + 3 * 262144, pWo_ft, 512, 512},
    {ft_w1, pW1_ft, 512, 2048}, {ft_w2, pW2_ft, 2048, 512},
    {fv_aw + 0 * 262144, pWq_fv, 512, 512}, {fv_aw + 1 * 262144, pWk_fv, 512, 512},
    {fv_aw + 2 * 262144, pWv_fv, 512, 512}, {fv_aw + 3 * 262144, pWo_fv, 512, 512},
    {fv_w1, pW1_fv, 512, 2048}, {fv_w2, pW2_fv, 2048, 512},
    {af_w + 0 * 262144, pW00, 512, 512}, {af_w + 1 * 262144, pW01, 512, 512},
    {af_w + 2 * 262144, pW10, 512, 512}, {af_w + 3 * 262144, pW11, 512, 512},
  };
  for (int i = 0; i < 16; ++i) {
    size_t tot = (size_t)packs[i].K * packs[i].N;
    pack_w_kernel<<<dim3((unsigned)((tot + 255) / 256)), 256, 0, stream>>>(
        packs[i].src, WP + packs[i].dst, packs[i].K, packs[i].N);
  }

  // vis -> padded
  {
    size_t tot = (size_t)NB * LVVALID * DMODEL;
    pad_copy_kernel<<<dim3((unsigned)((tot + 255) / 256)), 256, 0, stream>>>(
        vis, VISP, LVVALID, LPV, NB);
  }

  const int MV = NB * LPV;   // 7168
  const int MX = NB * LXX;   // 8192
  const float scMHA = 0.125f;
  const float scAF = 0.04419417382415922f;

  auto gemm = [&](const float* A, size_t wOff, const float* bias, const float* r1,
                  const float* r2, float* o, int M, int N, int K, int relu) {
    dim3 g(N / 128, M / 64);
    if (relu)
      gemm_kernel<1, 0, 0><<<g, 256, 0, stream>>>(A, WP + wOff, bias, r1, r2, o, M, N, K);
    else if (r1 && r2)
      gemm_kernel<0, 1, 1><<<g, 256, 0, stream>>>(A, WP + wOff, bias, r1, r2, o, M, N, K);
    else if (r1)
      gemm_kernel<0, 1, 0><<<g, 256, 0, stream>>>(A, WP + wOff, bias, r1, r2, o, M, N, K);
    else
      gemm_kernel<0, 0, 0><<<g, 256, 0, stream>>>(A, WP + wOff, bias, r1, r2, o, M, N, K);
  };
  auto score = [&](const float* Qp, const float* Kp, int Lq, int Lk, int hd, int H,
                   float sc) {
    score_kernel<<<dim3((Lk + 127) / 128, (Lq + 63) / 64, NB * H), 256, 0, stream>>>(
        Qp, Kp, Sb, Lq, Lk, hd, H, sc);
  };
  auto softm = [&](int Lk, int valid, int rows) {
    softmax_kernel<<<dim3((rows + 7) / 8), 256, 0, stream>>>(Sb, Pb, Lk, valid, rows);
  };
  auto packA = [&](const float* src, int rows) {
    size_t tot = (size_t)NB * rows * DMODEL;
    pack_act_kernel<<<dim3((unsigned)((tot + 255) / 256)), 256, 0, stream>>>(src, PKb,
                                                                             rows, NB);
  };
  auto pav = [&](float* o, int Lq, int Lk, int headW, int H) {
    pav_kernel<<<dim3((headW + 127) / 128, (Lq + 63) / 64, NB * H), 256, 0, stream>>>(
        Pb, PKb, o, Lq, Lk, headW, H);
  };
  auto ln = [&](const float* src, float* dst, const float* g, const float* bt, int rows) {
    ln_kernel<<<dim3((NB * rows + 7) / 8), 256, 0, stream>>>(src, dst, g, bt, rows, rows,
                                                             NB);
  };

  // ================= layer 1: cross(vis, text; ft) =====================
  ln(VISP, LNb, ft_ln + 0, ft_ln + 512, LPV);
  gemm(LNb, pWq_ft, ft_ab + 0, nullptr, nullptr, Qb, MV, 512, 512, 0);
  gemm(text, pWk_ft, ft_ab + 512, nullptr, nullptr, Kb, MX, 512, 512, 0);
  gemm(text, pWv_ft, ft_ab + 1024, nullptr, nullptr, Vb, MX, 512, 512, 0);
  score(Qb, Kb, LPV, 256, 64, 8, scMHA);
  softm(256, 256, NB * 8 * LPV);
  packA(Vb, 256);
  pav(LNb, LPV, 256, 64, 8);                                            // attn out
  gemm(LNb, pWo_ft, ft_ab + 1536, VISP, nullptr, Qb, MV, 512, 512, 0);  // m_a
  ln(Qb, LNb, ft_ln + 1024, ft_ln + 1536, LPV);
  gemm(LNb, pW1_ft, ft_b1, nullptr, nullptr, Sb, MV, 2048, 512, 1);     // relu ffn
  gemm(Sb, pW2_ft, ft_b2, Qb, nullptr, M1b, MV, 512, 2048, 0);          // m1

  // ================= layer 2: cross(m1, vis; fv) + m1 ==================
  ln(M1b, LNb, fv_ln + 0, fv_ln + 512, LPV);
  gemm(LNb, pWq_fv, fv_ab + 0, nullptr, nullptr, Qb, MV, 512, 512, 0);
  gemm(VISP, pWk_fv, fv_ab + 512, nullptr, nullptr, Kb, MV, 512, 512, 0);
  gemm(VISP, pWv_fv, fv_ab + 1024, nullptr, nullptr, Vb, MV, 512, 512, 0);
  score(Qb, Kb, LPV, LPV, 64, 8, scMHA);
  softm(LPV, LVVALID, NB * 8 * LPV);
  packA(Vb, LPV);
  pav(LNb, LPV, LPV, 64, 8);
  gemm(LNb, pWo_fv, fv_ab + 1536, M1b, nullptr, Qb, MV, 512, 512, 0);   // m_b
  ln(Qb, LNb, fv_ln + 1024, fv_ln + 1536, LPV);
  gemm(LNb, pW1_fv, fv_b1, nullptr, nullptr, Sb, MV, 2048, 512, 1);
  gemm(Sb, pW2_fv, fv_b2, Qb, M1b, Vb, MV, 512, 2048, 0);               // med_pre
  ln(Vb, M1b, ldm_ln, ldm_ln + 512, LPV);                               // med

  // ================= AF1: v = af(med, x, x) ============================
  gemm(M1b, pW00, af_b + 0, nullptr, nullptr, Qb, MV, 512, 512, 0);     // q = med@Wq
  score(Qb, x, LPV, 256, 512, 1, scAF);
  softm(256, 256, NB * LPV);
  packA(x, 256);
  pav(LNb, LPV, 256, 512, 1);                                           // o = P@x
  gemm(LNb, pW01, af_b + 512, nullptr, nullptr, Vb, MV, 512, 512, 0);   // vaf

  // ================= AF2: out = af(x, med, vaf) + x ====================
  gemm(x, pW10, af_b + 1024, nullptr, nullptr, Kb, MX, 512, 512, 0);    // q2 = x@Wq
  score(Kb, M1b, 256, LPV, 512, 1, scAF);
  softm(LPV, LVVALID, NB * 256);
  packA(Vb, LPV);
  pav(LNb, 256, LPV, 512, 1);                                           // o2 = P@vaf
  gemm(LNb, pW11, af_b + 1536, x, nullptr, outp, MX, 512, 512, 0);      // + x
}
